// Attention_79336635892513
// MI455X (gfx1250) — compile-verified, matrix operand order checked by reading, not run on hardware
//
#include <hip/hip_runtime.h>

// ---------------------------------------------------------------------------
// Causal GQA attention block for MI455X (gfx1250), wave32 WMMA bf16.
// B=1, S=2048, DIM=4096, NH=32, NKV=8, HD=128, WIN=4096 (>=S -> pure causal).
// GEMMs: V_WMMA_F32_16X16X32_BF16, weights pre-cast to bf16, W tiles staged
// into LDS via the Tensor Data Mover (double-buffered, TENSORcnt-synced).
// ---------------------------------------------------------------------------

#define S_LEN   2048
#define DIM     4096
#define NH      32
#define NKV     8
#define HD      128
#define SCALE   0.08838834764831845f   // 1/sqrt(128)
#define NEG_VAL (-1000000000.0f)

typedef __bf16    bf16;
typedef bf16      v16bf __attribute__((ext_vector_type(16)));
typedef float     v8f   __attribute__((ext_vector_type(8)));
typedef unsigned  uvec4 __attribute__((ext_vector_type(4)));
typedef unsigned  uvec2 __attribute__((ext_vector_type(2)));
typedef float     fvec4 __attribute__((ext_vector_type(4)));
typedef int       ivec8 __attribute__((ext_vector_type(8)));
typedef int       ivec4 __attribute__((ext_vector_type(4)));

#if defined(__has_builtin)
#if __has_builtin(__builtin_amdgcn_tensor_load_to_lds) && __has_builtin(__builtin_amdgcn_s_wait_tensorcnt)
#define USE_TDM 1
#endif
#endif

union FragU { v16bf v; uvec4 q[2]; };

// float -> bf16 bits, round-to-nearest-even
__device__ __forceinline__ unsigned short f2bf(float x) {
  unsigned u = __float_as_uint(x);
  u = (u + 0x7FFFu + ((u >> 16) & 1u)) >> 16;
  return (unsigned short)u;
}

// 16x32 bf16 WMMA fragment load (A-layout; B identical with lane=N):
// element i <-> k = (i&7) + 8*g + 16*(i>>3), g = lane>>4 -> two 16B loads.
__device__ __forceinline__ v16bf frag_load(const unsigned short* p, int g) {
  FragU f;
  f.q[0] = *reinterpret_cast<const uvec4*>(p + 8 * g);
  f.q[1] = *reinterpret_cast<const uvec4*>(p + 16 + 8 * g);
  return f.v;
}

__device__ __forceinline__ v8f wmma_bf16(v16bf a, v16bf b, v8f c) {
  return __builtin_amdgcn_wmma_f32_16x16x32_bf16(
      false, a, false, b, (short)0, c, false, false);
}

#ifdef USE_TDM
// TDM: DMA a 128-row x 32-col bf16 tile (row stride DIM elements) into LDS.
// D# bitfields per CDNA5 ISA ch.7/8 (group0: count/lds/global/type,
// group1: data_size=2B, tensor dims, tile 32x128, dim0 stride).
// This toolchain declares the 6-arg builtin:
//   (uint32x4 g0, int32x8 g1, int32x4 g2, int32x4 g3, int32x8 extra, i32 cpol)
__device__ __forceinline__ void tdm_load_w(unsigned lds_addr,
                                           const unsigned short* gptr,
                                           unsigned nrows) {
  unsigned long long ga = (unsigned long long)(uintptr_t)gptr;
  uvec4 g0;
  g0.x = 1u;                                   // count = 1 valid descriptor
  g0.y = lds_addr;                             // LDS byte address
  g0.z = (unsigned)(ga & 0xffffffffu);         // global_addr[31:0]
  g0.w = (unsigned)((ga >> 32) & 0x01ffffffu)  // global_addr[56:32]
       | (2u << 30);                           // type = 2 ("image")
  ivec8 g1;
  g1[0] = 0x00010000;                          // data_size = 1 (2 bytes)
  g1[1] = (int)(((unsigned)DIM & 0xffffu) << 16);        // tensor_dim0 lo16
  g1[2] = (int)(((unsigned)DIM >> 16) & 0xffffu)         // tensor_dim0 hi16
        | (int)((nrows & 0xffffu) << 16);                // tensor_dim1 lo16
  g1[3] = (int)((nrows >> 16) & 0xffffu)                 // tensor_dim1 hi16
        | (32 << 16);                                    // tile_dim0 = 32
  g1[4] = 128;                                 // tile_dim1 = 128, tile_dim2 = 0
  g1[5] = DIM;                                 // tensor_dim0_stride lo32
  g1[6] = 0;
  g1[7] = 0;
  ivec4 z4 = {0, 0, 0, 0};
  ivec8 z8 = {0, 0, 0, 0, 0, 0, 0, 0};
  __builtin_amdgcn_tensor_load_to_lds(g0, g1, z4, z4, z8, 0);
}
#endif

// ---------------------------------------------------------------------------
// Kernel: cast f32 array -> bf16 (4 elements / thread)
// ---------------------------------------------------------------------------
__global__ __launch_bounds__(256)
void k_cast(const float* __restrict__ x, unsigned short* __restrict__ xb, int n4) {
  int i = blockIdx.x * 256 + threadIdx.x;
  if (i >= n4) return;
  fvec4 v = reinterpret_cast<const fvec4*>(x)[i];
  uvec2 pk;
  pk.x = (unsigned)f2bf(v.x) | ((unsigned)f2bf(v.y) << 16);
  pk.y = (unsigned)f2bf(v.z) | ((unsigned)f2bf(v.w) << 16);
  reinterpret_cast<uvec2*>(xb)[i] = pk;
}

// ---------------------------------------------------------------------------
// GEMM + RoPE epilogue.  C(32 x 128) = A(32 x 4096, bf16) * Wb^T (bf16 rows).
// grid = (S/32, N/128), block = 128 (4 waves; wave w owns cols w*32..w*32+31,
// 2x2 sub-tiles -> 4 WMMAs per K-step).  W tile (128x32 bf16, 8KB) staged via
// TDM double-buffer (TENSORcnt) or vector-copy fallback.
// mode 0: Q  -> bout = Qb bf16 (S,4096), RoPE, pre-scaled by SCALE
// mode 1: K  -> fout = cache_k f32 + bout = Kb bf16 (kvh,S,HD), RoPE
// mode 2: V  -> fout = cache_v f32 + bout = Vt bf16 (kvh,HD,S), no RoPE
// mode 3: O  -> fout = d_out f32 (S,4096), plain
// ---------------------------------------------------------------------------
__global__ __launch_bounds__(128)
void k_gemm_rope(const unsigned short* __restrict__ A,
                 const unsigned short* __restrict__ Wb,
                 const float* __restrict__ fcos,
                 const float* __restrict__ fsin,
                 float* __restrict__ fout,
                 unsigned short* __restrict__ bout,
                 int nrows,     // rows of W (4096 or 1024)
                 int mode) {
  __shared__ __align__(16) unsigned short Wlds[2][128 * 32];  // 2 x 8KB bf16
  __shared__ __align__(16) float Ctile[32 * 128];             // 16KB epilogue

  const int tid  = threadIdx.x;
  const int w    = tid >> 5;
  const int lane = tid & 31;
  const int nl   = lane & 15;
  const int g    = lane >> 4;
  const int q0   = blockIdx.x * 32;
  const int wn0  = blockIdx.y * 128;

  v8f Cacc[2][2];
#pragma unroll
  for (int a = 0; a < 2; ++a)
#pragma unroll
    for (int b = 0; b < 2; ++b)
#pragma unroll
      for (int j = 0; j < 8; ++j) Cacc[a][b][j] = 0.0f;

  const unsigned short* a0 = A + (size_t)(q0 + nl) * DIM;
  const unsigned short* a1 = A + (size_t)(q0 + 16 + nl) * DIM;
  const unsigned short* wg = Wb + (size_t)wn0 * DIM;

#ifdef USE_TDM
  const unsigned ldsW0 = (unsigned)(uintptr_t)&Wlds[0][0];
  const unsigned ldsW1 = (unsigned)(uintptr_t)&Wlds[1][0];
  if (w == 0) tdm_load_w(ldsW0, wg, (unsigned)nrows);  // preload tile 0
#endif

  for (int it = 0; it < DIM / 32; ++it) {
    const int kc = it * 32;
#ifdef USE_TDM
    if (w == 0) __builtin_amdgcn_s_wait_tensorcnt(0);  // current tile landed
    __syncthreads();
    if (w == 0 && it + 1 < DIM / 32)                   // kick next DMA
      tdm_load_w((it & 1) ? ldsW0 : ldsW1, wg + kc + 32, (unsigned)nrows);
    const unsigned short* Wcur = &Wlds[it & 1][0];
#else
    {
      const unsigned short* src = wg + kc;
#pragma unroll
      for (int i = 0; i < 4; ++i) {
        int idx = tid * 4 + i;             // 512 x 16B = 8KB tile
        int row = idx >> 2, c8 = idx & 3;
        *reinterpret_cast<uvec4*>(&Wlds[0][row * 32 + c8 * 8]) =
            *reinterpret_cast<const uvec4*>(src + (size_t)row * DIM + c8 * 8);
      }
    }
    __syncthreads();
    const unsigned short* Wcur = &Wlds[0][0];
#endif
    __builtin_prefetch(a0 + kc + 64, 0, 3);            // global_prefetch_b8
    v16bf fa0 = frag_load(a0 + kc, g);
    v16bf fa1 = frag_load(a1 + kc, g);
    v16bf fb0 = frag_load(Wcur + (w * 32 + nl) * 32, g);
    v16bf fb1 = frag_load(Wcur + (w * 32 + 16 + nl) * 32, g);
    Cacc[0][0] = wmma_bf16(fa0, fb0, Cacc[0][0]);
    Cacc[0][1] = wmma_bf16(fa0, fb1, Cacc[0][1]);
    Cacc[1][0] = wmma_bf16(fa1, fb0, Cacc[1][0]);
    Cacc[1][1] = wmma_bf16(fa1, fb1, Cacc[1][1]);
    __syncthreads();
  }

  if (mode == 3) {  // plain output projection -> f32
#pragma unroll
    for (int rr = 0; rr < 2; ++rr)
#pragma unroll
      for (int cc = 0; cc < 2; ++cc)
#pragma unroll
        for (int r = 0; r < 8; ++r)
          fout[(size_t)(q0 + rr * 16 + r + 8 * g) * DIM +
               wn0 + w * 32 + cc * 16 + nl] = Cacc[rr][cc][r];
    return;
  }

  // Stage C through LDS so RoPE pairs (adjacent columns) are thread-local
#pragma unroll
  for (int rr = 0; rr < 2; ++rr)
#pragma unroll
    for (int cc = 0; cc < 2; ++cc)
#pragma unroll
      for (int r = 0; r < 8; ++r)
        Ctile[(rr * 16 + r + 8 * g) * 128 + w * 32 + cc * 16 + nl] = Cacc[rr][cc][r];
  __syncthreads();

  for (int p = tid; p < 2048; p += 128) {  // 32 rows * 64 pairs
    int row = p >> 6, pc = p & 63;
    int s = q0 + row;
    int n = wn0 + 2 * pc;                  // even global column
    int d = n & (HD - 1);
    float xr = Ctile[row * 128 + 2 * pc];
    float xi = Ctile[row * 128 + 2 * pc + 1];
    float c = 1.0f, sn = 0.0f;
    if (mode != 2) {
      c  = fcos[(size_t)s * (HD / 2) + (d >> 1)];
      sn = fsin[(size_t)s * (HD / 2) + (d >> 1)];
    }
    float our = xr * c - xi * sn;
    float oui = xr * sn + xi * c;
    if (mode == 0) {                       // Q: bf16, fold in 1/sqrt(HD)
      bout[(size_t)s * DIM + n]     = f2bf(our * SCALE);
      bout[(size_t)s * DIM + n + 1] = f2bf(oui * SCALE);
    } else {
      fout[(size_t)s * (NKV * HD) + n]     = our;      // cache (f32)
      fout[(size_t)s * (NKV * HD) + n + 1] = oui;
      int kvh = n >> 7;
      if (mode == 1) {                     // K bf16: (kvh, S, HD)
        bout[(size_t)kvh * S_LEN * HD + (size_t)s * HD + d]     = f2bf(our);
        bout[(size_t)kvh * S_LEN * HD + (size_t)s * HD + d + 1] = f2bf(oui);
      } else {                             // V bf16 transposed: (kvh, HD, S)
        bout[(size_t)kvh * HD * S_LEN + (size_t)d * S_LEN + s]       = f2bf(xr);
        bout[(size_t)kvh * HD * S_LEN + (size_t)(d + 1) * S_LEN + s] = f2bf(xi);
      }
    }
  }
}

// ---------------------------------------------------------------------------
// Copy untouched cache tail (rows S..WIN-1) from input caches to outputs.
// ---------------------------------------------------------------------------
__global__ __launch_bounds__(256)
void k_cache_tail(const float* __restrict__ ck, const float* __restrict__ cv,
                  float* __restrict__ ock, float* __restrict__ ocv, int tail) {
  int i = blockIdx.x * 256 + threadIdx.x;
  if (i >= tail) return;
  if (blockIdx.y == 0) ock[tail + i] = ck[tail + i];
  else                 ocv[tail + i] = cv[tail + i];
}

// ---------------------------------------------------------------------------
// Flash attention: one wave per (head, 16-query tile).  grid = (NH, S/16).
// Q pre-scaled bf16; K (kvh,S,HD) bf16; V^T (kvh,HD,S) bf16; out bf16 (S,4096).
// ---------------------------------------------------------------------------
__global__ __launch_bounds__(32)
void k_attn(const unsigned short* __restrict__ Qb,
            const unsigned short* __restrict__ Kb,
            const unsigned short* __restrict__ Vt,
            unsigned short* __restrict__ attb) {
  const int h    = blockIdx.x;
  const int q0   = blockIdx.y * 16;
  const int kvh  = h >> 2;                 // GQA: 4 Q heads per KV head
  const int lane = threadIdx.x;
  const int nl   = lane & 15;
  const int g    = lane >> 4;

  __shared__ __align__(16) unsigned short Plds[16 * 32];

  v16bf qf[4];
  const unsigned short* qbase = Qb + (size_t)(q0 + nl) * DIM + h * HD;
#pragma unroll
  for (int c = 0; c < 4; ++c) qf[c] = frag_load(qbase + c * 32, g);

  v8f O[8];
  float m[8], l[8];
#pragma unroll
  for (int t = 0; t < 8; ++t)
#pragma unroll
    for (int j = 0; j < 8; ++j) O[t][j] = 0.0f;
#pragma unroll
  for (int r = 0; r < 8; ++r) { m[r] = -__builtin_inff(); l[r] = 0.0f; }

  const unsigned short* kbase = Kb + (size_t)kvh * S_LEN * HD;
  const unsigned short* vbase = Vt + (size_t)kvh * HD * S_LEN;
  const int qmax = q0 + 15;

  for (int kv0 = 0; kv0 <= qmax; kv0 += 32) {   // S % 32 == 0 -> in bounds
    v8f Ca, Cb;
#pragma unroll
    for (int j = 0; j < 8; ++j) { Ca[j] = 0.0f; Cb[j] = 0.0f; }

    const unsigned short* ka = kbase + (size_t)(kv0 + nl) * HD;
    const unsigned short* kb = kbase + (size_t)(kv0 + 16 + nl) * HD;
#pragma unroll
    for (int c = 0; c < 4; ++c) {
      Ca = wmma_bf16(qf[c], frag_load(ka + c * 32, g), Ca);
      Cb = wmma_bf16(qf[c], frag_load(kb + c * 32, g), Cb);
    }

    // Causal mask + online softmax (rows M = r + 8*g live across 16 lanes)
#pragma unroll
    for (int r = 0; r < 8; ++r) {
      int M = q0 + r + 8 * g;
      if (kv0 + nl > M)      Ca[r] += NEG_VAL;
      if (kv0 + 16 + nl > M) Cb[r] += NEG_VAL;

      float mx = fmaxf(Ca[r], Cb[r]);
#pragma unroll
      for (int off = 1; off < 16; off <<= 1)
        mx = fmaxf(mx, __shfl_xor(mx, off, 32));
      float mn   = fmaxf(m[r], mx);
      float corr = __expf(m[r] - mn);
      float pa   = __expf(Ca[r] - mn);
      float pb   = __expf(Cb[r] - mn);
      float rs   = pa + pb;
#pragma unroll
      for (int off = 1; off < 16; off <<= 1)
        rs += __shfl_xor(rs, off, 32);
      l[r] = l[r] * corr + rs;
      m[r] = mn;
#pragma unroll
      for (int t = 0; t < 8; ++t) O[t][r] *= corr;

      Plds[(r + 8 * g) * 32 + nl]      = f2bf(pa);
      Plds[(r + 8 * g) * 32 + 16 + nl] = f2bf(pb);
    }
    __syncthreads();

    v16bf pf = frag_load(&Plds[nl * 32], g);  // P as A fragment (16x32)
#pragma unroll
    for (int t = 0; t < 8; ++t) {
      const unsigned short* vb = vbase + (size_t)(t * 16 + nl) * S_LEN + kv0;
      O[t] = wmma_bf16(pf, frag_load(vb, g), O[t]);
    }
    __syncthreads();
  }

#pragma unroll
  for (int r = 0; r < 8; ++r) {
    float inv = 1.0f / l[r];
    size_t rowbase = (size_t)(q0 + r + 8 * g) * DIM + h * HD;
#pragma unroll
    for (int t = 0; t < 8; ++t)
      attb[rowbase + t * 16 + nl] = f2bf(O[t][r] * inv);
  }
}

// ---------------------------------------------------------------------------
// Host-side launch
// ---------------------------------------------------------------------------
extern "C" void kernel_launch(void* const* d_in, const int* in_sizes, int n_in,
                              void* d_out, int out_size, void* d_ws, size_t ws_size,
                              hipStream_t stream) {
  (void)in_sizes; (void)n_in; (void)out_size; (void)ws_size;

  const float* x     = (const float*)d_in[0];
  const float* fcos  = (const float*)d_in[1];
  const float* fsin  = (const float*)d_in[2];
  // d_in[3]=positions, d_in[4]=mask: implied by causal indexing (WIN>=S)
  const float* wq    = (const float*)d_in[5];
  const float* wk    = (const float*)d_in[6];
  const float* wv    = (const float*)d_in[7];
  const float* wo    = (const float*)d_in[8];
  const float* ck_in = (const float*)d_in[9];
  const float* cv_in = (const float*)d_in[10];

  float* out     = (float*)d_out;                 // (2048, 4096)
  float* cache_k = out + (size_t)S_LEN * DIM;     // (4096, 8, 128)
  float* cache_v = cache_k + (size_t)4096 * NKV * HD;

  char* ws = (char*)d_ws;
  unsigned short* xb   = (unsigned short*)(ws);                    // 16 MB
  unsigned short* qb   = (unsigned short*)(ws + 16777216ull);      // 16 MB
  unsigned short* kb   = (unsigned short*)(ws + 33554432ull);      //  4 MB
  unsigned short* vtb  = (unsigned short*)(ws + 37748736ull);      //  4 MB
  unsigned short* attb = (unsigned short*)(ws + 41943040ull);      // 16 MB
  unsigned short* wqb  = (unsigned short*)(ws + 58720256ull);      // 32 MB
  unsigned short* wkb  = (unsigned short*)(ws + 92274688ull);      //  8 MB
  unsigned short* wvb  = (unsigned short*)(ws + 100663296ull);     //  8 MB
  unsigned short* wob  = (unsigned short*)(ws + 109051904ull);     // 32 MB

  // 1) one-time bf16 casts (activations + weights)
  k_cast<<<(2097152 + 255) / 256, 256, 0, stream>>>(x,  xb,  2097152);
  k_cast<<<(4194304 + 255) / 256, 256, 0, stream>>>(wq, wqb, 4194304);
  k_cast<<<(1048576 + 255) / 256, 256, 0, stream>>>(wk, wkb, 1048576);
  k_cast<<<(1048576 + 255) / 256, 256, 0, stream>>>(wv, wvb, 1048576);
  k_cast<<<(4194304 + 255) / 256, 256, 0, stream>>>(wo, wob, 4194304);

  // 2) Q/K/V projections (+RoPE, +cache writes, +bf16 staging)
  k_gemm_rope<<<dim3(S_LEN / 32, DIM / 128), 128, 0, stream>>>(
      xb, wqb, fcos, fsin, nullptr, qb, DIM, 0);
  k_gemm_rope<<<dim3(S_LEN / 32, (NKV * HD) / 128), 128, 0, stream>>>(
      xb, wkb, fcos, fsin, cache_k, kb, NKV * HD, 1);
  k_gemm_rope<<<dim3(S_LEN / 32, (NKV * HD) / 128), 128, 0, stream>>>(
      xb, wvb, fcos, fsin, cache_v, vtb, NKV * HD, 2);

  // 3) untouched cache tail (rows 2048..4095) copied from input caches
  {
    int tail = (4096 - S_LEN) * NKV * HD;  // 2,097,152 per tensor
    k_cache_tail<<<dim3((tail + 255) / 256, 2), 256, 0, stream>>>(
        ck_in, cv_in, cache_k, cache_v, tail);
  }

  // 4) flash attention (causal; WIN=4096 >= S so window mask is a no-op)
  k_attn<<<dim3(NH, S_LEN / 16), 32, 0, stream>>>(qb, kb, vtb, attb);

  // 5) output projection -> f32 result
  k_gemm_rope<<<dim3(S_LEN / 32, DIM / 128), 128, 0, stream>>>(
      attb, wob, fcos, fsin, out, nullptr, DIM, 3);
}
